// Mamba3Block_71768903516842
// MI455X (gfx1250) — matrix operand order repaired and checked by reading, not compiled, not run on hardware
//
#include <hip/hip_runtime.h>
#include <hip/hip_bf16.h>

typedef __attribute__((ext_vector_type(16))) __bf16 v16bf;
typedef __attribute__((ext_vector_type(8)))  __bf16 v8bf;
typedef __attribute__((ext_vector_type(8)))  float  v8f;

namespace {

constexpr int kL = 2048, kDM = 512, kH = 16, kG = 4, kP = 64, kN = 32;
constexpr int kChunk = 256, kNC = 8;
constexpr int kDProj = 2568, kDProjP = 2576;
constexpr int cZ = 0, cX = 1024, cB = 2048, cC = 2304, cDT = 2560, cLAM = 2564;
constexpr float kEps = 1e-5f;

__device__ __forceinline__ v8f wmma_bf16(v16bf a, v16bf b, v8f c) {
  return __builtin_amdgcn_wmma_f32_16x16x32_bf16(false, a, false, b, (short)0, c, false, false);
}
__device__ __forceinline__ v8f zero8f() {
  v8f z;
#pragma unroll
  for (int i = 0; i < 8; ++i) z[i] = 0.f;
  return z;
}
__device__ __forceinline__ v16bf cat8(v8bf a, v8bf b) {
  return __builtin_shufflevector(a, b, 0,1,2,3,4,5,6,7,8,9,10,11,12,13,14,15);
}
// A-operand (16xK, row-major, 16-bit): lane holds row (lane&15);
// vector elems = K{k0+hi*8..+7} ++ K{k0+16+hi*8..+7}  (CDNA5 ISA A layout)
__device__ __forceinline__ v16bf ldA(const __bf16* base, int stride, int row, int k0, int hi) {
  const v8bf* p0 = (const v8bf*)(base + (size_t)row * stride + k0 + hi * 8);
  const v8bf* p1 = (const v8bf*)(base + (size_t)row * stride + k0 + 16 + hi * 8);
  return cat8(*p0, *p1);
}
// B-operand (Kx16, source stored column-major [col][k]): lane holds column (lane&15),
// vector elems = K{k0+hi*8..+7} ++ K{k0+16+hi*8..+7}  (matches A layout)
__device__ __forceinline__ v16bf ldB(const __bf16* base, int stride, int col, int k0, int hi) {
  const v8bf* p0 = (const v8bf*)(base + (size_t)col * stride + k0 + hi * 8);
  const v8bf* p1 = (const v8bf*)(base + (size_t)col * stride + k0 + 16 + hi * 8);
  return cat8(*p0, *p1);
}
__device__ __forceinline__ v16bf ldA_f32(const float* base, int stride, int row, int k0, int hi) {
  const float* p0 = base + (size_t)row * stride + k0 + hi * 8;
  const float* p1 = p0 + 16;
  v16bf r;
#pragma unroll
  for (int i = 0; i < 8; ++i) { r[i] = (__bf16)p0[i]; r[8 + i] = (__bf16)p1[i]; }
  return r;
}
__device__ __forceinline__ float sigmoidf_(float x) { return 1.f / (1.f + __expf(-x)); }

// CDNA5 async global->LDS copy, 16B per lane, tracked by ASYNCcnt.
// LDS dest address = low 32 bits of the generic pointer (wave-relative LDS offset).
__device__ __forceinline__ void async_copy_b128(void* lds_ptr, const void* gptr) {
  unsigned lds_addr = (unsigned)(uintptr_t)lds_ptr;
  asm volatile("global_load_async_to_lds_b128 %0, %1, off"
               :: "v"(lds_addr), "v"(gptr)
               : "memory");
}

// ---------------------------------------------------------------- elementwise cvt
__global__ void cvt_bf16_kernel(const float* __restrict__ in, __bf16* __restrict__ out, int n) {
  int i = blockIdx.x * blockDim.x + threadIdx.x;
  if (i < n) out[i] = (__bf16)in[i];
}

// ---------------------------------------------------------------- generic WMMA GEMM
// C[MxN](f32) = A[MxK](bf16,lda) @ B[NxK]^T(bf16,ldb) + bias ; M%128==0, K%32==0
// Double-buffered: GLOBAL_LOAD_ASYNC_TO_LDS_B128 staging overlapped with WMMA.
__global__ __launch_bounds__(256) void gemm_bf16_kernel(
    const __bf16* __restrict__ A, int lda,
    const __bf16* __restrict__ B, int ldb, int Nact,
    const float* __restrict__ bias,
    float* __restrict__ C, int ldc, int K) {
  __shared__ __bf16 As[2][128 * 32];
  __shared__ __bf16 Bs[2][128 * 32];
  const int t = threadIdx.x, w = t >> 5, lane = t & 31, hi = lane >> 4, ln = lane & 15;
  const int wm = w >> 2, wn = w & 3;
  const long M0 = (long)blockIdx.y * 128;
  const int N0 = blockIdx.x * 128;
  v8f acc[4][2];
#pragma unroll
  for (int i = 0; i < 4; ++i)
#pragma unroll
    for (int j = 0; j < 2; ++j) acc[i][j] = zero8f();

  // per-stage async staging: 512 16B segments per tile, 2 per thread per tile,
  // exactly 4 async issues per thread per stage (uniform -> exact ASYNCcnt math).
  auto issue_stage = [&](int s) {
    const int k0 = s * 32, buf = s & 1;
#pragma unroll
    for (int q = 0; q < 2; ++q) {
      int cidx = t + q * 256;
      int rw = cidx >> 2, sub = cidx & 3;
      async_copy_b128(&As[buf][rw * 32 + sub * 8], &A[(M0 + rw) * lda + k0 + sub * 8]);
      int gn = N0 + rw;
      const __bf16* bp = (gn < Nact) ? &B[(size_t)gn * ldb + k0 + sub * 8] : B;
      async_copy_b128(&Bs[buf][rw * 32 + sub * 8], bp);
    }
  };

  const int S = K / 32;
  issue_stage(0);
  for (int s = 0; s < S; ++s) {
    const int buf = s & 1;
    if (s + 1 < S) {
      issue_stage(s + 1);
      // drain everything except the 4 just-issued (in-order completion)
      asm volatile("s_wait_asynccnt 0x4" ::: "memory");
    } else {
      asm volatile("s_wait_asynccnt 0x0" ::: "memory");
    }
    __syncthreads();
    v16bf af[4], bfv[2];
#pragma unroll
    for (int it = 0; it < 4; ++it) af[it] = ldA(As[buf], 32, wm * 64 + it * 16 + ln, 0, hi);
#pragma unroll
    for (int nt = 0; nt < 2; ++nt) bfv[nt] = ldB(Bs[buf], 32, wn * 32 + nt * 16 + ln, 0, hi);
#pragma unroll
    for (int it = 0; it < 4; ++it)
#pragma unroll
      for (int nt = 0; nt < 2; ++nt) acc[it][nt] = wmma_bf16(af[it], bfv[nt], acc[it][nt]);
    __syncthreads();  // all waves done reading buf before stage s+2 overwrites it
  }
#pragma unroll
  for (int it = 0; it < 4; ++it)
#pragma unroll
    for (int nt = 0; nt < 2; ++nt)
#pragma unroll
      for (int v = 0; v < 8; ++v) {
        long m = M0 + wm * 64 + it * 16 + v + hi * 8;
        int n = N0 + wn * 32 + nt * 16 + ln;
        if (n < Nact) C[m * ldc + n] = acc[it][nt][v] + (bias ? bias[n] : 0.f);
      }
}

// ---------------------------------------------------------------- dt / lambda scan (tiny)
__global__ void scan_kernel(const float* __restrict__ proj, const float* __restrict__ A_log,
                            float* __restrict__ dtg, float* __restrict__ cumdt,
                            float* __restrict__ cumA, float* __restrict__ lamg) {
  int gi = threadIdx.x;
  if (gi >= kG) return;
  const float Ag = -__expf(A_log[gi]);
  float cd = 0.f, ca = 0.f;
  for (int l = 0; l < kL; ++l) {
    float p = proj[(size_t)l * kDProjP + cDT + gi];
    float dt = (p > 20.f) ? p : log1pf(__expf(p));
    cd += dt; ca += dt * Ag;
    dtg[l * 4 + gi] = dt;
    cumdt[l * 4 + gi] = cd;
    cumA[l * 4 + gi] = ca;
    lamg[l * 4 + gi] = sigmoidf_(proj[(size_t)l * kDProjP + cLAM + gi]);
  }
}

// ---------------------------------------------------------------- RMS + RoPE on B/C, silu(x)
__global__ __launch_bounds__(256) void bcx_kernel(
    const float* __restrict__ proj, const float* __restrict__ theta_log,
    const float* __restrict__ cumdt,
    const float* __restrict__ normB, const float* __restrict__ normC,
    const float* __restrict__ biasB, const float* __restrict__ biasC,
    __bf16* __restrict__ Brot, __bf16* __restrict__ Crot, __bf16* __restrict__ sx) {
  const int l = blockIdx.x, t = threadIdx.x;
  const int g = t >> 6, rem = t & 63;
  __shared__ float vals[256], nv[256], msr[4];
  const float* row = proj + (size_t)l * kDProjP;
  const int n2 = rem >> 2, which = (t >> 1) & 1, nn = rem >> 1, r = rem & 1;
  const float ang = __expf(theta_log[g * 16 + n2]) * cumdt[l * 4 + g];
  const float ca = __cosf(ang), sa = __sinf(ang);

  // ---- B: rms -> +bias -> rope(-ang) ; store [l][r][n]
  float x = row[cB + t];
  vals[t] = x * x;
  __syncthreads();
  if (t < 4) {
    float s = 0.f;
    for (int i = 0; i < 64; ++i) s += vals[t * 64 + i];
    msr[t] = rsqrtf(s * (1.f / 64.f) + kEps);
  }
  __syncthreads();
  nv[t] = x * msr[g] * normB[rem] + biasB[t];
  __syncthreads();
  {
    float re = nv[t & ~2], im = nv[t | 2];
    float outv = (which == 0) ? (re * ca + im * sa) : (-re * sa + im * ca);
    Brot[(size_t)l * 64 + r * 32 + nn] = (__bf16)outv;
  }
  __syncthreads();
  // ---- C: rope(+ang)
  x = row[cC + t];
  vals[t] = x * x;
  __syncthreads();
  if (t < 4) {
    float s = 0.f;
    for (int i = 0; i < 64; ++i) s += vals[t * 64 + i];
    msr[t] = rsqrtf(s * (1.f / 64.f) + kEps);
  }
  __syncthreads();
  nv[t] = x * msr[g] * normC[rem] + biasC[t];
  __syncthreads();
  {
    float re = nv[t & ~2], im = nv[t | 2];
    float outv = (which == 0) ? (re * ca - im * sa) : (re * sa + im * ca);
    Crot[(size_t)l * 64 + r * 32 + nn] = (__bf16)outv;
  }
  // ---- silu(xp)
#pragma unroll
  for (int q = 0; q < 4; ++q) {
    int idx = q * 256 + t;
    float xv = row[cX + idx];
    sx[(size_t)l * 1024 + idx] = (__bf16)(xv * sigmoidf_(xv));
  }
}

// ---------------------------------------------------------------- x_up = silu(xp) @ W_xup^T
__global__ __launch_bounds__(256) void xup_kernel(
    const __bf16* __restrict__ sx, const __bf16* __restrict__ Wxup,
    __bf16* __restrict__ xup) {
  const int t = threadIdx.x, w = t >> 5, lane = t & 31, hi = lane >> 4, ln = lane & 15;
  const int m0 = blockIdx.x * 128 + w * 16;
  v8f acc[8];
#pragma unroll
  for (int nt = 0; nt < 8; ++nt) acc[nt] = zero8f();
  for (int k0 = 0; k0 < 64; k0 += 32) {
    v16bf af = ldA(sx, 64, m0 + ln, k0, hi);
#pragma unroll
    for (int nt = 0; nt < 8; ++nt)
      acc[nt] = wmma_bf16(af, ldB(Wxup, 64, nt * 16 + ln, k0, hi), acc[nt]);
  }
#pragma unroll
  for (int nt = 0; nt < 8; ++nt)
#pragma unroll
    for (int v = 0; v < 8; ++v) {
      int m = m0 + v + hi * 8;
      xup[(size_t)m * 128 + nt * 16 + ln] = (__bf16)acc[nt][v];
    }
}

// ---------------------------------------------------------------- intra-chunk y_diag (SSD form)
// per (chunk c, head h): G = C_r @ B_r'^T (one WMMA per 16x16 tile, K=32=n),
// scale by c_w[i,j] = dt_j*exp(cumA_i-cumA_j)*((1-lam_j)[i>=j] + lam_{j+1}[i>j]),
// then (c_w o G) @ x_r' via WMMA; also accumulate last-row state partial.
__global__ __launch_bounds__(256) void ydiag_kernel(
    const __bf16* __restrict__ Brot, const __bf16* __restrict__ Crot,
    const __bf16* __restrict__ xup,
    const float* __restrict__ cumA, const float* __restrict__ dtg,
    const float* __restrict__ lamg,
    float* __restrict__ ybuf, float* __restrict__ hlast) {
  const int c = blockIdx.x, h = blockIdx.y, g = h >> 2;
  const int t = threadIdx.x, w = t >> 5, lane = t & 31, hi = lane >> 4, ln = lane & 15;
  const int l0 = c * kChunk;
  __shared__ __bf16 xT[64 * 256];      // [p][j] for current r'
  __shared__ __bf16 Ms[8][16 * 32];    // per-wave (c_w o G) staging
  __shared__ __bf16 Bl[256 * 32];      // [j][n] for current r'
  __shared__ float cums_s[256], dt_s[256], lam_s[257], cwl[256];
  {
    int l = l0 + t;
    cums_s[t] = cumA[l * 4 + g];
    dt_s[t] = dtg[l * 4 + g];
    lam_s[t] = lamg[l * 4 + g];
    if (t == 0) lam_s[256] = 0.f;
  }
  __syncthreads();
  {
    float nxt = (t < 255) ? lam_s[t + 1] : 0.f;
    cwl[t] = dt_s[t] * __expf(cums_s[255] - cums_s[t]) * ((1.f - lam_s[t]) + nxt);
  }

  v8f acc[2][2][4];
#pragma unroll
  for (int a = 0; a < 2; ++a)
#pragma unroll
    for (int b = 0; b < 2; ++b)
#pragma unroll
      for (int pp = 0; pp < 4; ++pp) acc[a][b][pp] = zero8f();
  float hacc[8];
#pragma unroll
  for (int q = 0; q < 8; ++q) hacc[q] = 0.f;
  const int hn = t & 31, hp0 = (t >> 5) * 8;

  for (int rp = 0; rp < 2; ++rp) {
    __syncthreads();
    for (int kk = 0; kk < 64; ++kk) {          // xT[p][j] <- xup[l0+j][h][p*2+rp]
      int idx = kk * 256 + t;
      int p = idx & 63, j = idx >> 6;
      xT[p * 256 + j] = xup[((size_t)(l0 + j) * 16 + h) * 128 + p * 2 + rp];
    }
    for (int kk = 0; kk < 32; ++kk) {          // Bl[j][n]
      int idx = kk * 256 + t;
      int j = idx >> 5, n1 = idx & 31;
      Bl[idx] = Brot[(size_t)(l0 + j) * 64 + rp * 32 + n1];
    }
    __syncthreads();
    for (int it = 0; it < 2; ++it) {
      const int i0 = (w * 2 + it) * 16;
      for (int r = 0; r < 2; ++r) {
        v16bf cf = ldA(Crot, 64, l0 + i0 + ln, r * 32, hi);
        for (int kt = 0; kt < 8; ++kt) {
#pragma unroll
          for (int jh = 0; jh < 2; ++jh) {
            const int jj = kt * 32 + jh * 16;
            v16bf bfB = ldB(Bl, 32, jj + ln, 0, hi);           // B-op col=j, K=n
            v8f Gt = wmma_bf16(cf, bfB, zero8f());             // G[i,j] = C_r . B_r'
            const int jloc = jj + ln;
            const float dtj = dt_s[jloc], l1 = lam_s[jloc], l2 = lam_s[jloc + 1];
            const float cj = cums_s[jloc];
#pragma unroll
            for (int v = 0; v < 8; ++v) {
              const int iloc = i0 + v + hi * 8;
              float cw = 0.f;
              if (iloc >= jloc) {
                float wt = (1.f - l1) + ((iloc > jloc && jloc < 255) ? l2 : 0.f);
                cw = dtj * __expf(cums_s[iloc] - cj) * wt;
              }
              Ms[w][(v + hi * 8) * 32 + jh * 16 + ln] = (__bf16)(Gt[v] * cw);
            }
          }
          v16bf af = ldA(&Ms[w][0], 32, ln, 0, hi);            // (c_w o G) as A-op
#pragma unroll
          for (int pt = 0; pt < 4; ++pt) {
            v16bf bx = ldB(xT, 256, pt * 16 + ln, kt * 32, hi);
            acc[it][r][pt] = wmma_bf16(af, bx, acc[it][r][pt]);
          }
        }
      }
    }
    // last-row state partial: hlast_part[n,p] += sum_j cwl[j]*B[j,n]*x[j,p]
    for (int j = 0; j < 256; ++j) {
      float cb = cwl[j] * (float)Bl[j * 32 + hn];
#pragma unroll
      for (int q = 0; q < 8; ++q) hacc[q] += cb * (float)xT[(hp0 + q) * 256 + j];
    }
  }
#pragma unroll
  for (int it = 0; it < 2; ++it)
#pragma unroll
    for (int r = 0; r < 2; ++r)
#pragma unroll
      for (int pt = 0; pt < 4; ++pt)
#pragma unroll
        for (int v = 0; v < 8; ++v) {
          int iloc = (w * 2 + it) * 16 + v + hi * 8;
          int l = l0 + iloc, p = pt * 16 + ln;
          ybuf[((size_t)l * 16 + h) * 128 + p * 2 + r] = acc[it][r][pt][v];
        }
#pragma unroll
  for (int q = 0; q < 8; ++q)
    hlast[(((size_t)c * 16 + h) * 32 + hn) * 64 + hp0 + q] = hacc[q];
}

// ---------------------------------------------------------------- inter-chunk state scan
__global__ __launch_bounds__(256) void state_scan_kernel(
    const __bf16* __restrict__ Brot, const __bf16* __restrict__ xup,
    const float* __restrict__ dtg, const float* __restrict__ lamg,
    const float* __restrict__ cumA, const float* __restrict__ hlast,
    float* __restrict__ seff) {
  const int h = blockIdx.y, g = h >> 2;
  const int pair = blockIdx.x * 256 + threadIdx.x;
  const int nn = pair >> 6, p = pair & 63;
  float st = 0.f;
  for (int c = 0; c < kNC; ++c) {
    const int l0 = c * kChunk;
    float se = st;
    if (c > 0) {  // boundary shift term folds into the state: + lam_first * v[first-1]
      const int jb = l0 - 1;
      float x0 = (float)xup[((size_t)jb * 16 + h) * 128 + p * 2 + 0];
      float x1 = (float)xup[((size_t)jb * 16 + h) * 128 + p * 2 + 1];
      float b0 = (float)Brot[(size_t)jb * 64 + nn];
      float b1 = (float)Brot[(size_t)jb * 64 + 32 + nn];
      se += lamg[l0 * 4 + g] * dtg[jb * 4 + g] * (b0 * x0 + b1 * x1);
    }
    const size_t idx = (((size_t)c * 16 + h) * 32 + nn) * 64 + p;
    seff[idx] = se;
    float decay = __expf(cumA[(l0 + 255) * 4 + g] - ((c > 0) ? cumA[(l0 - 1) * 4 + g] : 0.f));
    st = decay * se + hlast[idx];
  }
}

// ---------------------------------------------------------------- y_off = exp(cums_i) * C @ state
__global__ __launch_bounds__(256) void yoff_kernel(
    const __bf16* __restrict__ Crot, const float* __restrict__ seff,
    const float* __restrict__ cumA, float* __restrict__ ybuf) {
  const int c = blockIdx.x, h = blockIdx.y, g = h >> 2;
  const int t = threadIdx.x, w = t >> 5, lane = t & 31, hi = lane >> 4, ln = lane & 15;
  const int l0 = c * kChunk;
  __shared__ __bf16 sT[64 * 32];   // state^T [p][n]
  __shared__ float E[256];
#pragma unroll
  for (int q = 0; q < 8; ++q) {
    int idx = q * 256 + t;
    int nn = idx >> 6, p = idx & 63;
    sT[p * 32 + nn] = (__bf16)seff[(((size_t)c * 16 + h) * 32 + nn) * 64 + p];
  }
  float basev = (c > 0) ? cumA[(size_t)(l0 - 1) * 4 + g] : 0.f;
  E[t] = __expf(cumA[(size_t)(l0 + t) * 4 + g] - basev);
  __syncthreads();
  for (int it = 0; it < 2; ++it) {
    const int i0 = (w * 2 + it) * 16;
    for (int r = 0; r < 2; ++r) {
      v16bf cf = ldA(Crot, 64, l0 + i0 + ln, r * 32, hi);
#pragma unroll
      for (int pt = 0; pt < 4; ++pt) {
        v16bf sf = ldB(sT, 32, pt * 16 + ln, 0, hi);
        v8f d = wmma_bf16(cf, sf, zero8f());
#pragma unroll
        for (int v = 0; v < 8; ++v) {
          int iloc = i0 + v + hi * 8;
          int l = l0 + iloc, p = pt * 16 + ln;
          size_t yi = ((size_t)l * 16 + h) * 128 + p * 2 + r;
          ybuf[yi] += d[v] * E[iloc];
        }
      }
    }
  }
}

// ---------------------------------------------------------------- down-proj + D-skip + z-gate
__global__ __launch_bounds__(256) void down_gate_kernel(
    const float* __restrict__ ybuf, const __bf16* __restrict__ Wyd,
    const __bf16* __restrict__ sx, const float* __restrict__ proj,
    const float* __restrict__ Dvec, __bf16* __restrict__ gbuf) {
  const int t = threadIdx.x, w = t >> 5, lane = t & 31, hi = lane >> 4, ln = lane & 15;
  const int m0 = blockIdx.x * 128 + w * 16;
  v8f acc[4];
#pragma unroll
  for (int nt = 0; nt < 4; ++nt) acc[nt] = zero8f();
  for (int k0 = 0; k0 < 128; k0 += 32) {
    v16bf af = ldA_f32(ybuf, 128, m0 + ln, k0, hi);
#pragma unroll
    for (int nt = 0; nt < 4; ++nt)
      acc[nt] = wmma_bf16(af, ldB(Wyd, 128, nt * 16 + ln, k0, hi), acc[nt]);
  }
#pragma unroll
  for (int nt = 0; nt < 4; ++nt)
#pragma unroll
    for (int v = 0; v < 8; ++v) {
      int m = m0 + v + hi * 8;
      int n = nt * 16 + ln;
      int l = m >> 4, hh = m & 15;
      float val = acc[nt][v] + Dvec[hh] * (float)sx[(size_t)m * 64 + n];
      float zv = proj[(size_t)l * kDProjP + cZ + hh * 64 + n];
      gbuf[(size_t)m * 64 + n] = (__bf16)(val * zv * sigmoidf_(zv));
    }
}

}  // namespace

extern "C" void kernel_launch(void* const* d_in, const int* in_sizes, int n_in,
                              void* d_out, int out_size, void* d_ws, size_t ws_size,
                              hipStream_t stream) {
  (void)in_sizes; (void)n_in; (void)out_size; (void)ws_size;
  const float* u         = (const float*)d_in[0];
  const float* W_in      = (const float*)d_in[1];
  const float* b_in      = (const float*)d_in[2];
  const float* W_xup     = (const float*)d_in[3];
  const float* W_ydown   = (const float*)d_in[4];
  const float* A_log     = (const float*)d_in[5];
  const float* theta_log = (const float*)d_in[6];
  const float* Dvec      = (const float*)d_in[7];
  const float* norm_B_w  = (const float*)d_in[8];
  const float* norm_C_w  = (const float*)d_in[9];
  const float* bias_B    = (const float*)d_in[10];
  const float* bias_C    = (const float*)d_in[11];
  const float* W_out     = (const float*)d_in[12];
  float* out = (float*)d_out;

  char* wsb = (char*)d_ws;
  size_t off = 0;
  auto carve = [&](size_t bytes) -> void* {
    void* p = wsb + off;
    off += (bytes + 255) & ~(size_t)255;
    return p;
  };
  __bf16* u_bf    = (__bf16*)carve((size_t)kL * kDM * 2);
  __bf16* win_bf  = (__bf16*)carve((size_t)kDProj * kDM * 2);
  __bf16* wxup_bf = (__bf16*)carve((size_t)128 * 64 * 2);
  __bf16* wyd_bf  = (__bf16*)carve((size_t)64 * 128 * 2);
  __bf16* wout_bf = (__bf16*)carve((size_t)kDM * 1024 * 2);
  float*  proj    = (float*)carve((size_t)kL * kDProjP * 4);
  float*  dtg     = (float*)carve((size_t)kL * kG * 4);
  float*  cumdt   = (float*)carve((size_t)kL * kG * 4);
  float*  cumA    = (float*)carve((size_t)kL * kG * 4);
  float*  lamg    = (float*)carve((size_t)kL * kG * 4);
  __bf16* Brot    = (__bf16*)carve((size_t)kL * 64 * 2);
  __bf16* Crot    = (__bf16*)carve((size_t)kL * 64 * 2);
  __bf16* sxb     = (__bf16*)carve((size_t)kL * 1024 * 2);
  __bf16* xupb    = (__bf16*)carve((size_t)kL * kH * 128 * 2);
  float*  ybuf    = (float*)carve((size_t)kL * kH * 128 * 4);
  float*  hlastb  = (float*)carve((size_t)kNC * kH * kN * kP * 4);
  float*  seffb   = (float*)carve((size_t)kNC * kH * kN * kP * 4);
  __bf16* gbuf    = (__bf16*)carve((size_t)kL * 1024 * 2);

  auto cvt = [&](const float* src, __bf16* dst, size_t n) {
    int blocks = (int)((n + 255) / 256);
    cvt_bf16_kernel<<<blocks, 256, 0, stream>>>(src, dst, (int)n);
  };
  cvt(u, u_bf, (size_t)kL * kDM);
  cvt(W_in, win_bf, (size_t)kDProj * kDM);
  cvt(W_xup, wxup_bf, 128 * 64);
  cvt(W_ydown, wyd_bf, 64 * 128);
  cvt(W_out, wout_bf, (size_t)kDM * 1024);

  // proj = u @ W_in^T + b_in   (2048 x 2568, padded ldc 2576)
  gemm_bf16_kernel<<<dim3((kDProj + 127) / 128, kL / 128), 256, 0, stream>>>(
      u_bf, kDM, win_bf, kDM, kDProj, b_in, proj, kDProjP, kDM);
  scan_kernel<<<1, 64, 0, stream>>>(proj, A_log, dtg, cumdt, cumA, lamg);
  bcx_kernel<<<kL, 256, 0, stream>>>(proj, theta_log, cumdt, norm_B_w, norm_C_w,
                                     bias_B, bias_C, Brot, Crot, sxb);
  xup_kernel<<<(kL * kH) / 128, 256, 0, stream>>>(sxb, wxup_bf, xupb);
  ydiag_kernel<<<dim3(kNC, kH), 256, 0, stream>>>(Brot, Crot, xupb, cumA, dtg, lamg,
                                                  ybuf, hlastb);
  state_scan_kernel<<<dim3(8, kH), 256, 0, stream>>>(Brot, xupb, dtg, lamg, cumA,
                                                     hlastb, seffb);
  yoff_kernel<<<dim3(kNC, kH), 256, 0, stream>>>(Crot, seffb, cumA, ybuf);
  down_gate_kernel<<<(kL * kH) / 128, 256, 0, stream>>>(ybuf, wyd_bf, sxb, proj, Dvec, gbuf);
  // out = g @ W_out^T  (2048 x 512)
  gemm_bf16_kernel<<<dim3(kDM / 128, kL / 128), 256, 0, stream>>>(
      gbuf, 1024, wout_bf, 1024, kDM, nullptr, out, kDM, 1024);
}
